// LSTM_28948079575638
// MI455X (gfx1250) — matrix operand order repaired and checked by reading, not compiled, not run on hardware
//
#include <hip/hip_runtime.h>
#include <hip/hip_bf16.h>

// ---------------------------------------------------------------------------
// LSTM on MI455X (gfx1250): bf16 WMMA (16x16x32, f32 accum) for the input
// projection and recurrent GEMM, fused with the cell nonlinearity.
// - Weights pre-packed once into chunk-major bf16 panels.
// - B panels staged into LDS with GLOBAL_LOAD_ASYNC_TO_LDS_B128 (ASYNCcnt),
//   double-buffered: async fill of chunk ci+1 overlaps the WMMAs of chunk ci,
//   drained with s_wait_asynccnt right before the publishing barrier.
// - A fragments software-pipelined in registers one chunk ahead.
// - 512 stream-ordered step launches express the scan dependency; h is
//   double-buffered in bf16.
// ---------------------------------------------------------------------------

#define T_STEPS 512
#define BATCH   128
#define EMB     512
#define HID     1024
#define G4      (4 * HID)
#define OUTP    2

// K-chunk staged in LDS per double-buffer slot
#define KC         64
#define NC_EMB     (EMB / KC)          // 8  chunks from w_ih
#define NC_HID     (HID / KC)          // 16 chunks from w_hh
#define NCHUNK     (NC_EMB + NC_HID)   // 24
#define ROWS       64                  // 4 gates x 16 n-columns
#define CHUNK_ELTS (ROWS * KC)         // 4096 bf16 = 8KB
#define LDS_STRIDE (KC + 8)            // 36-dword row stride: conflict-free b128

typedef __attribute__((ext_vector_type(16))) __bf16 v16bf;
typedef __attribute__((ext_vector_type(8)))  __bf16 v8bf;
typedef __attribute__((ext_vector_type(8)))  float  v8f;

// ---- small helpers --------------------------------------------------------

__device__ __forceinline__ float sigmoid_fast(float x) {
    return 1.0f / (1.0f + __expf(-x));
}
__device__ __forceinline__ float tanh_fast(float x) {
    return 2.0f / (1.0f + __expf(-2.0f * x)) - 1.0f;   // 2*sigmoid(2x)-1
}

__device__ __forceinline__ v16bf cat8(v8bf lo, v8bf hi) {
    v16bf a;
#pragma unroll
    for (int j = 0; j < 8; ++j) { a[j] = lo[j]; a[j + 8] = hi[j]; }
    return a;
}

__device__ __forceinline__ v16bf cvt_a(v8f e0, v8f e1) {
    v16bf a;
#pragma unroll
    for (int j = 0; j < 8; ++j) { a[j] = (__bf16)e0[j]; a[j + 8] = (__bf16)e1[j]; }
    return a;
}

// ---- prep kernels ---------------------------------------------------------

// Pack w_ih / w_hh (f32) into chunk-major bf16 panels:
//   wpack[ntile][ci][r][k], r = gate*16 + n, k in [0,KC)
// so each (ntile, ci) block is one contiguous 8KB LDS fill.
__global__ __launch_bounds__(256) void pack_weights(
    const float* __restrict__ w_ih, const float* __restrict__ w_hh,
    __bf16* __restrict__ wpack, long ntotal) {
    long o = (long)blockIdx.x * blockDim.x + threadIdx.x;
    if (o >= ntotal) return;
    const int  k     = (int)(o % KC);
    long q           = o / KC;
    const int  r     = (int)(q % ROWS);
    q /= ROWS;
    const int  ci    = (int)(q % NCHUNK);
    const int  ntile = (int)(q / NCHUNK);
    const int  g = r >> 4, n = r & 15;
    const int  row = g * HID + ntile * 16 + n;
    float v;
    if (ci < NC_EMB) v = w_ih[(long)row * EMB + ci * KC + k];
    else             v = w_hh[(long)row * HID + (ci - NC_EMB) * KC + k];
    wpack[o] = (__bf16)v;
}

__global__ __launch_bounds__(256) void bias_sum(const float* __restrict__ a,
                                                const float* __restrict__ b,
                                                float* __restrict__ d, int n) {
    int i = blockIdx.x * blockDim.x + threadIdx.x;
    if (i < n) d[i] = a[i] + b[i];
}

__global__ __launch_bounds__(256) void zero_state(__bf16* __restrict__ h0,
                                                  __bf16* __restrict__ h1,
                                                  float* __restrict__ c, int n) {
    int i = blockIdx.x * blockDim.x + threadIdx.x;
    if (i < n) {
        h0[i] = (__bf16)0.0f;
        h1[i] = (__bf16)0.0f;
        c[i]  = 0.0f;
    }
}

// ---- recurrent step kernel ------------------------------------------------
//
// grid.x = HID/16 = 64 n-tiles, block = 256 threads = 8 waves (wave32).
// Wave w owns the 16(batch) x 16(hidden) tile and accumulates 4 gate tiles
// with v_wmma_f32_16x16x32_bf16 over K = EMB (embedding gather, f32->bf16
// in-register) + K = HID (previous h, bf16).
//
// Per chunk (KC=64 K-values): all 8 waves share one LDS-staged B panel.
// The panel is filled with GLOBAL_LOAD_ASYNC_TO_LDS_B128 (no VGPR round-trip,
// ASYNCcnt-tracked) and double-buffered; A fragments are pipelined in
// registers one chunk ahead, so steady-state WMMAs wait only on ds_loads.
//
// Fragment layouts (CDNA5 ISA 7.12.2), lane = half*16 + l:
//   A (16x32):  row M=l;  elems j<8 -> K=k0+half*8+j ; j>=8 -> K=k0+16+half*8+(j-8)
//   B (32x16):  col N=l;  elems j<8 -> K=k0+half*8+j ; j>=8 -> K=k0+16+half*8+(j-8)
//   C/D (16x16): col N=l; vgpr r -> row M=half*8+r

// Async memory->LDS fill of one contiguous 8KB chunk into padded LDS rows.
// INST_OFFSET is added to BOTH the global and LDS addresses (ISA 10.7), so a
// single base covers each thread's 32B slice via offset:0 / offset:16.
__device__ __forceinline__ void async_fill_chunk(__bf16* __restrict__ dst,
                                                 const __bf16* __restrict__ src,
                                                 int tid) {
    const int r = tid >> 2;          // destination row 0..63
    const int p = tid & 3;           // 32B piece within the row
    const __bf16* g = src + (size_t)tid * 16;
    // Low 32 bits of the flat shared-aperture address are the LDS byte offset.
    const unsigned d = (unsigned)(uintptr_t)(dst + r * LDS_STRIDE + p * 16);
    asm volatile("global_load_async_to_lds_b128 %0, %1, off"
                 :: "v"(d), "v"(g) : "memory");
    asm volatile("global_load_async_to_lds_b128 %0, %1, off offset:16"
                 :: "v"(d), "v"(g) : "memory");
}

__device__ __forceinline__ void wait_async_done() {
    asm volatile("s_wait_asynccnt 0x0" ::: "memory");
}

__device__ __forceinline__ v16bf load_a(int ci, int kk,
                                        const float* __restrict__ erow,
                                        const __bf16* __restrict__ hrow,
                                        int half) {
    if (ci < NC_EMB) {                                   // uniform branch
        const int k0 = ci * KC + kk;
        v8f e0 = *(const v8f*)(erow + k0 + half * 8);
        v8f e1 = *(const v8f*)(erow + k0 + 16 + half * 8);
        return cvt_a(e0, e1);
    } else {
        const int k0 = (ci - NC_EMB) * KC + kk;
        return cat8(*(const v8bf*)(hrow + k0 + half * 8),
                    *(const v8bf*)(hrow + k0 + 16 + half * 8));
    }
}

__global__ __launch_bounds__(256) void lstm_step(
    const int*    __restrict__ x,      // [T, B]
    const float*  __restrict__ emb,    // [VOCAB, EMB] f32
    const __bf16* __restrict__ wpack,  // packed panels, see pack_weights
    const float*  __restrict__ bsum,   // [4H] = b_ih + b_hh
    const __bf16* __restrict__ h_in,   // [B, HID] bf16
    __bf16*       __restrict__ h_out,  // [B, HID] bf16
    float*        __restrict__ c,      // [B, HID] f32
    int t)
{
    __shared__ __bf16 Bsm[2][ROWS * LDS_STRIDE];   // 2 x 9KB double buffer

    const int tid  = threadIdx.x;
    const int lane = tid & 31;
    const int wave = tid >> 5;
    const int l    = lane & 15;
    const int half = lane >> 4;

    const int m0 = wave * 16;        // batch tile base
    const int n0 = blockIdx.x * 16;  // hidden-column tile base

    v8f acc[4];
#pragma unroll
    for (int g = 0; g < 4; ++g) acc[g] = (v8f){0.f,0.f,0.f,0.f,0.f,0.f,0.f,0.f};

    const int  b_row = m0 + l;
    const long idx   = (long)x[(long)t * BATCH + b_row];
    const float*  erow = emb + idx * (long)EMB;
    const __bf16* hrow = h_in + (long)b_row * HID;
    const __bf16* wp   = wpack + (long)blockIdx.x * ((long)NCHUNK * CHUNK_ELTS);

    // Prologue: async-stage chunk 0 and load its A fragments.
    async_fill_chunk(&Bsm[0][0], wp, tid);
    v16bf a0 = load_a(0, 0,  erow, hrow, half);
    v16bf a1 = load_a(0, 32, erow, hrow, half);
    wait_async_done();
    __syncthreads();

    for (int ci = 0; ci < NCHUNK; ++ci) {
        const int cur = ci & 1;

        // Issue next chunk's async LDS fill and A-fragment loads before
        // computing, so their latency hides behind this chunk's 8 WMMAs.
        v16bf an0 = a0, an1 = a1;
        if (ci + 1 < NCHUNK) {
            async_fill_chunk(&Bsm[cur ^ 1][0], wp + (size_t)(ci + 1) * CHUNK_ELTS, tid);
            an0 = load_a(ci + 1, 0,  erow, hrow, half);
            an1 = load_a(ci + 1, 32, erow, hrow, half);
        }

        const __bf16* bpan = &Bsm[cur][0];
#pragma unroll
        for (int g = 0; g < 4; ++g) {
            const __bf16* bp = bpan + (g * 16 + l) * LDS_STRIDE + half * 8;
            v16bf b0 = cat8(*(const v8bf*)bp, *(const v8bf*)(bp + 16));
            acc[g] = __builtin_amdgcn_wmma_f32_16x16x32_bf16(
                false, a0, false, b0, (short)0, acc[g], false, false);
            v16bf b1 = cat8(*(const v8bf*)(bp + 32), *(const v8bf*)(bp + 48));
            acc[g] = __builtin_amdgcn_wmma_f32_16x16x32_bf16(
                false, a1, false, b1, (short)0, acc[g], false, false);
        }

        a0 = an0;
        a1 = an1;
        wait_async_done();   // our async fills of buf[cur^1] have landed
        __syncthreads();     // buf[cur] reusable; buf[cur^1] published
    }

    // ---- Cell update on D layout: lane holds N=n0+l, vgpr r -> M=m0+half*8+r ----
    const int n = n0 + l;
    const float bi = bsum[0 * HID + n];
    const float bf = bsum[1 * HID + n];
    const float bg = bsum[2 * HID + n];
    const float bo = bsum[3 * HID + n];

#pragma unroll
    for (int r = 0; r < 8; ++r) {
        const int  m   = m0 + half * 8 + r;
        const long oix = (long)m * HID + n;
        const float gi = sigmoid_fast(acc[0][r] + bi);
        const float gf = sigmoid_fast(acc[1][r] + bf);
        const float gg = tanh_fast   (acc[2][r] + bg);
        const float go = sigmoid_fast(acc[3][r] + bo);
        const float cc = gf * c[oix] + gi * gg;
        c[oix]     = cc;
        h_out[oix] = (__bf16)(go * tanh_fast(cc));
    }
}

// ---- final FC head --------------------------------------------------------

__global__ __launch_bounds__(256) void fc_head(const __bf16* __restrict__ h,
                                               const float* __restrict__ w,
                                               const float* __restrict__ bias,
                                               float* __restrict__ out) {
    const int tid = threadIdx.x;           // 256 threads = 128 batch x 2 outputs
    const int b = tid >> 1;
    const int o = tid & 1;
    float s = bias[o];
    for (int n = 0; n < HID; ++n)
        s += (float)h[(long)b * HID + n] * w[o * HID + n];
    out[b * OUTP + o] = s;
}

// ---------------------------------------------------------------------------

extern "C" void kernel_launch(void* const* d_in, const int* in_sizes, int n_in,
                              void* d_out, int out_size, void* d_ws, size_t ws_size,
                              hipStream_t stream) {
    const int*   x    = (const int*)  d_in[0];
    const float* emb  = (const float*)d_in[1];
    const float* w_ih = (const float*)d_in[2];
    const float* w_hh = (const float*)d_in[3];
    const float* b_ih = (const float*)d_in[4];
    const float* b_hh = (const float*)d_in[5];
    const float* fc_w = (const float*)d_in[6];
    const float* fc_b = (const float*)d_in[7];
    float* out = (float*)d_out;

    // Workspace layout
    char* ws = (char*)d_ws;
    size_t off = 0;
    const long npack = (long)(HID / 16) * NCHUNK * CHUNK_ELTS;   // 4096*1536 elts
    __bf16* wpack = (__bf16*)(ws + off); off += (size_t)npack * sizeof(__bf16); // 12 MB
    float*  bsum  = (float*) (ws + off); off += (size_t)G4 * sizeof(float);     // 16 KB
    __bf16* h0    = (__bf16*)(ws + off); off += (size_t)BATCH * HID * sizeof(__bf16);
    __bf16* h1    = (__bf16*)(ws + off); off += (size_t)BATCH * HID * sizeof(__bf16);
    float*  cst   = (float*) (ws + off); off += (size_t)BATCH * HID * sizeof(float);
    (void)ws_size; (void)in_sizes; (void)n_in; (void)out_size;

    // One-time prep: pack+convert weights, fold biases, zero state.
    pack_weights<<<(int)((npack + 255) / 256), 256, 0, stream>>>(w_ih, w_hh, wpack, npack);
    bias_sum<<<(G4 + 255) / 256, 256, 0, stream>>>(b_ih, b_hh, bsum, G4);
    {
        int ns = BATCH * HID;
        zero_state<<<(ns + 255) / 256, 256, 0, stream>>>(h0, h1, cst, ns);
    }

    // Recurrence: 512 stream-ordered step kernels, h double-buffered.
    __bf16* hbuf[2] = { h0, h1 };
    for (int t = 0; t < T_STEPS; ++t) {
        lstm_step<<<HID / 16, 256, 0, stream>>>(
            x, emb, wpack, bsum,
            hbuf[t & 1], hbuf[(t + 1) & 1], cst, t);
    }

    // Final projection: h after 512 steps lives in hbuf[512 & 1] == hbuf[0]
    fc_head<<<1, 256, 0, stream>>>(hbuf[T_STEPS & 1], fc_w, fc_b, out);
}
